// Model_81020263071765
// MI455X (gfx1250) — compile-verified, matrix operand order checked by reading, not run on hardware
//
#include <hip/hip_runtime.h>
#include <hip/hip_bf16.h>

// ---------------------------------------------------------------------------
// CDNA5 (gfx1250) GNN pipeline using v_wmma_f32_16x16x32_f16.
// Dims padded to multiples of 16 with zero weights (exact zeros via relu).
// ---------------------------------------------------------------------------

typedef __attribute__((ext_vector_type(16))) _Float16 v16h;
typedef __attribute__((ext_vector_type(8)))  _Float16 v8h;
typedef __attribute__((ext_vector_type(8)))  float    v8f;
typedef __attribute__((ext_vector_type(4)))  float    v4f;

#define N_NODES 50000
#define N_EDGES 1600000
#define N_VMS   10000
#define N_SCOREK 1000000

// ---- WMMA helpers ---------------------------------------------------------

static __device__ __forceinline__ v8f wmma16(v16h a, v16h b, v8f c) {
  // D = A(16x32 f16) * B(32x16 f16) + C(16x16 f32)
  return __builtin_amdgcn_wmma_f32_16x16x32_f16(
      false, a, false, b, (short)0, c, false, false);
}

// A-fragment (16x32 f16) from row-major f16 row pointer.
// ISA layout: lanes 0-15 hold K = k0..k0+7 and k0+16..k0+23,
//             lanes 16-31 hold K = k0+8..k0+15 and k0+24..k0+31  (hi = lane>>4)
static __device__ __forceinline__ v16h frag_f16(const _Float16* row, int k0, int hi) {
  v8h lo = *(const v8h*)(row + k0 + hi * 8);
  v8h hh = *(const v8h*)(row + k0 + 16 + hi * 8);
  v16h a;
#pragma unroll
  for (int i = 0; i < 8; ++i) { a[i] = lo[i]; a[i + 8] = hh[i]; }
  return a;
}

// Same fragment, but source is f32 (converted on the fly).
static __device__ __forceinline__ v16h frag_f32(const float* row, int k0, int hi) {
  v4f x0 = *(const v4f*)(row + k0 + hi * 8);
  v4f x1 = *(const v4f*)(row + k0 + hi * 8 + 4);
  v4f y0 = *(const v4f*)(row + k0 + 16 + hi * 8);
  v4f y1 = *(const v4f*)(row + k0 + 16 + hi * 8 + 4);
  v16h a;
#pragma unroll
  for (int i = 0; i < 4; ++i) {
    a[i]      = (_Float16)x0[i];
    a[i + 4]  = (_Float16)x1[i];
    a[i + 8]  = (_Float16)y0[i];
    a[i + 12] = (_Float16)y1[i];
  }
  return a;
}

static __device__ __forceinline__ void atomicAddF(float* p, float v) {
  __hip_atomic_fetch_add(p, v, __ATOMIC_RELAXED, __HIP_MEMORY_SCOPE_AGENT);
}

// ---- weight packing: f32 [K,OUT] -> padded f16 [OUTp][Ktot] (B^T layout) ---

__global__ void pack_w(const float* __restrict__ src, _Float16* __restrict__ dst,
                       int KA_pad, int KA_real, int KB_real, int Ktot,
                       int OUTp, int OUTr) {
  int i = blockIdx.x * blockDim.x + threadIdx.x;
  int total = OUTp * Ktot;
  if (i >= total) return;
  int out = i / Ktot, k = i % Ktot;
  int srow = -1;
  if (k < KA_pad) { if (k < KA_real) srow = k; }
  else            { int j = k - KA_pad; if (j < KB_real) srow = KA_real + j; }
  float v = 0.f;
  if (srow >= 0 && out < OUTr) v = src[srow * OUTr + out];
  dst[i] = (_Float16)v;
}

__global__ void cvt_f32_f16(const float* __restrict__ s, _Float16* __restrict__ d, int n) {
  int i = blockIdx.x * blockDim.x + threadIdx.x;
  if (i < n) d[i] = (_Float16)s[i];
}

// ---- edge message + scatter-sum -------------------------------------------
// m = relu(h[src] @ Wm[:DH] + ef @ Wm[DH:] + bm);  neigh[dst] += m  (f32 atomics)
template <int DH, int DOUTP, int DOUTR>
__global__ __launch_bounds__(256) void edge_msg(
    const _Float16* __restrict__ hf, const float* __restrict__ ef,
    const int* __restrict__ src, const int* __restrict__ dst,
    const _Float16* __restrict__ wt_g, const float* __restrict__ bias,
    float* __restrict__ neigh, int nTiles) {
  constexpr int KT = DH + 64;
  __shared__ _Float16 wt[DOUTP * KT];
  for (int i = threadIdx.x; i < DOUTP * KT / 8; i += blockDim.x)
    ((v8h*)wt)[i] = ((const v8h*)wt_g)[i];
  __syncthreads();

  int wave = threadIdx.x >> 5, lane = threadIdx.x & 31;
  int tile = blockIdx.x * 8 + wave;
  if (tile >= nTiles) return;
  int r = lane & 15, hi = lane >> 4;
  int e0 = tile * 16;

  int srow = src[e0 + r];
  const _Float16* hrow = hf + (size_t)srow * DH;
  const float*    erow = ef + (size_t)(e0 + r) * 64;

  v8f c[DOUTP / 16];
#pragma unroll
  for (int t = 0; t < DOUTP / 16; ++t) c[t] = (v8f){};

#pragma unroll
  for (int kc = 0; kc < KT; kc += 32) {
    v16h a = (kc < DH) ? frag_f16(hrow, kc, hi) : frag_f32(erow, kc - DH, hi);
#pragma unroll
    for (int t = 0; t < DOUTP / 16; ++t) {
      v16h b = frag_f16(wt + (size_t)(t * 16 + r) * KT, kc, hi);
      c[t] = wmma16(a, b, c[t]);
    }
  }

  int drow[8];
#pragma unroll
  for (int v = 0; v < 8; ++v) drow[v] = dst[e0 + hi * 8 + v];

#pragma unroll
  for (int t = 0; t < DOUTP / 16; ++t) {
    int n = t * 16 + r;
    float bv = (n < DOUTR) ? bias[n] : 0.f;
#pragma unroll
    for (int v = 0; v < 8; ++v) {
      float val = c[t][v] + bv;
      val = val > 0.f ? val : 0.f;
      atomicAddF(&neigh[(size_t)drow[v] * DOUTP + n], val);
    }
  }
}

// ---- node apply -----------------------------------------------------------
// hnew = relu(h @ Wa[:DIN] + neigh @ Wa[DIN:] + ba); optional row-sum output.
template <int DIN, int DNB, int DOUTP, int DOUTR, bool STORE, bool SUM>
__global__ __launch_bounds__(256) void apply_k(
    const _Float16* __restrict__ hf, const float* __restrict__ neigh,
    const _Float16* __restrict__ wt_g, const float* __restrict__ bias,
    _Float16* __restrict__ hout, float* __restrict__ rowsum, int nTiles) {
  constexpr int KT = DIN + DNB;
  __shared__ _Float16 wt[DOUTP * KT];
  for (int i = threadIdx.x; i < DOUTP * KT / 8; i += blockDim.x)
    ((v8h*)wt)[i] = ((const v8h*)wt_g)[i];
  __syncthreads();

  int wave = threadIdx.x >> 5, lane = threadIdx.x & 31;
  int tile = blockIdx.x * 8 + wave;
  if (tile >= nTiles) return;
  int r = lane & 15, hi = lane >> 4;
  int node = tile * 16 + r;

  const _Float16* hrow = hf + (size_t)node * DIN;
  const float*    nrow = neigh + (size_t)node * DNB;

  v8f c[DOUTP / 16];
#pragma unroll
  for (int t = 0; t < DOUTP / 16; ++t) c[t] = (v8f){};

#pragma unroll
  for (int kc = 0; kc < KT; kc += 32) {
    v16h a = (kc < DIN) ? frag_f16(hrow, kc, hi) : frag_f32(nrow, kc - DIN, hi);
#pragma unroll
    for (int t = 0; t < DOUTP / 16; ++t) {
      v16h b = frag_f16(wt + (size_t)(t * 16 + r) * KT, kc, hi);
      c[t] = wmma16(a, b, c[t]);
    }
  }

  float s[8];
#pragma unroll
  for (int v = 0; v < 8; ++v) s[v] = 0.f;

#pragma unroll
  for (int t = 0; t < DOUTP / 16; ++t) {
    int n = t * 16 + r;
    float bv = (n < DOUTR) ? bias[n] : 0.f;
#pragma unroll
    for (int v = 0; v < 8; ++v) {
      float val = c[t][v] + bv;
      val = val > 0.f ? val : 0.f;
      if (STORE) {
        int m = tile * 16 + hi * 8 + v;
        hout[(size_t)m * DOUTP + n] = (_Float16)val;
      }
      if (SUM) s[v] += val;
    }
  }

  if (SUM) {
    // reduce across the 16 columns held by each half-wave
#pragma unroll
    for (int v = 0; v < 8; ++v)
      for (int m = 1; m < 16; m <<= 1) s[v] += __shfl_xor(s[v], m, 32);
    if (r == 0) {
#pragma unroll
      for (int v = 0; v < 8; ++v)
        rowsum[tile * 16 + hi * 8 + v] = s[v];
    }
  }
}

// ---- VM MLP tower + row-sum -----------------------------------------------
__global__ __launch_bounds__(128) void mlp_k(
    const float* __restrict__ vm, const _Float16* __restrict__ w1_g,
    const float* __restrict__ b1, const _Float16* __restrict__ w2_g,
    const float* __restrict__ b2, float* __restrict__ vmsum, int nTiles) {
  __shared__ _Float16 w1[128 * 64];   // [out=128][K=64]
  __shared__ _Float16 w2[64 * 128];   // [out=64][K=128]
  __shared__ _Float16 hid[4][16 * 128];
  for (int i = threadIdx.x; i < 128 * 64 / 8; i += blockDim.x)
    ((v8h*)w1)[i] = ((const v8h*)w1_g)[i];
  for (int i = threadIdx.x; i < 64 * 128 / 8; i += blockDim.x)
    ((v8h*)w2)[i] = ((const v8h*)w2_g)[i];
  __syncthreads();

  int wave = threadIdx.x >> 5, lane = threadIdx.x & 31;
  int tile = blockIdx.x * 4 + wave;
  if (tile >= nTiles) return;
  int r = lane & 15, hi = lane >> 4;

  const float* vrow = vm + (size_t)(tile * 16 + r) * 64;

  v8f c1[8];
#pragma unroll
  for (int t = 0; t < 8; ++t) c1[t] = (v8f){};
#pragma unroll
  for (int kc = 0; kc < 64; kc += 32) {
    v16h a = frag_f32(vrow, kc, hi);
#pragma unroll
    for (int t = 0; t < 8; ++t) {
      v16h b = frag_f16(w1 + (size_t)(t * 16 + r) * 64, kc, hi);
      c1[t] = wmma16(a, b, c1[t]);
    }
  }
  _Float16* myhid = hid[wave];
#pragma unroll
  for (int t = 0; t < 8; ++t) {
    int n = t * 16 + r;
    float bv = b1[n];
#pragma unroll
    for (int v = 0; v < 8; ++v) {
      float val = c1[t][v] + bv;
      val = val > 0.f ? val : 0.f;
      myhid[(size_t)(hi * 8 + v) * 128 + n] = (_Float16)val;
    }
  }

  v8f c2[4];
#pragma unroll
  for (int t = 0; t < 4; ++t) c2[t] = (v8f){};
#pragma unroll
  for (int kc = 0; kc < 128; kc += 32) {
    v16h a = frag_f16(myhid + (size_t)r * 128, kc, hi);
#pragma unroll
    for (int t = 0; t < 4; ++t) {
      v16h b = frag_f16(w2 + (size_t)(t * 16 + r) * 128, kc, hi);
      c2[t] = wmma16(a, b, c2[t]);
    }
  }

  float s[8];
#pragma unroll
  for (int v = 0; v < 8; ++v) s[v] = 0.f;
#pragma unroll
  for (int t = 0; t < 4; ++t) {
    int n = t * 16 + r;
    float bv = b2[n];
#pragma unroll
    for (int v = 0; v < 8; ++v) s[v] += c2[t][v] + bv;   // no relu on layer 2
  }
#pragma unroll
  for (int v = 0; v < 8; ++v)
    for (int m = 1; m < 16; m <<= 1) s[v] += __shfl_xor(s[v], m, 32);
  if (r == 0) {
#pragma unroll
    for (int v = 0; v < 8; ++v)
      vmsum[tile * 16 + hi * 8 + v] = s[v];
  }
}

// ---- final scoring --------------------------------------------------------
__global__ void score_k(const float* __restrict__ cs, const float* __restrict__ vs,
                        const int* __restrict__ ec, const int* __restrict__ ev,
                        float* __restrict__ out, int n) {
  int i = blockIdx.x * blockDim.x + threadIdx.x;
  if (i < n) {
    float x = cs[ec[i]] + vs[ev[i]];
    out[i] = 1.f / (1.f + __expf(-x));
  }
}

// ---------------------------------------------------------------------------

extern "C" void kernel_launch(void* const* d_in, const int* in_sizes, int n_in,
                              void* d_out, int out_size, void* d_ws, size_t ws_size,
                              hipStream_t stream) {
  const float* comp = (const float*)d_in[0];
  const float* ef   = (const float*)d_in[1];
  const float* vm   = (const float*)d_in[2];
  const int* src    = (const int*)d_in[3];
  const int* dst    = (const int*)d_in[4];
  const int* ecomp  = (const int*)d_in[5];
  const int* evm    = (const int*)d_in[6];
  const float* Wm1 = (const float*)d_in[7];  const float* bm1 = (const float*)d_in[8];
  const float* Wa1 = (const float*)d_in[9];  const float* ba1 = (const float*)d_in[10];
  const float* Wm2 = (const float*)d_in[11]; const float* bm2 = (const float*)d_in[12];
  const float* Wa2 = (const float*)d_in[13]; const float* ba2 = (const float*)d_in[14];
  const float* Wm3 = (const float*)d_in[15]; const float* bm3 = (const float*)d_in[16];
  const float* Wa3 = (const float*)d_in[17]; const float* ba3 = (const float*)d_in[18];
  const float* W1  = (const float*)d_in[19]; const float* b1  = (const float*)d_in[20];
  const float* W2  = (const float*)d_in[21]; const float* b2  = (const float*)d_in[22];

  (void)in_sizes; (void)n_in; (void)ws_size;

  char* ws = (char*)d_ws;
  size_t off = 0;
  auto alloc = [&](size_t bytes) -> char* {
    char* p = ws + off;
    off = (off + bytes + 255) & ~(size_t)255;
    return p;
  };

  _Float16* wm1t = (_Float16*)alloc(64 * 192 * 2);
  _Float16* wa1t = (_Float16*)alloc(64 * 192 * 2);
  _Float16* wm2t = (_Float16*)alloc(32 * 128 * 2);
  _Float16* wa2t = (_Float16*)alloc(32 * 96 * 2);
  _Float16* wm3t = (_Float16*)alloc(64 * 96 * 2);
  _Float16* wa3t = (_Float16*)alloc(64 * 96 * 2);
  _Float16* w1t  = (_Float16*)alloc(128 * 64 * 2);
  _Float16* w2t  = (_Float16*)alloc(64 * 128 * 2);
  _Float16* h0f  = (_Float16*)alloc((size_t)N_NODES * 128 * 2);
  _Float16* h1f  = (_Float16*)alloc((size_t)N_NODES * 64 * 2);
  _Float16* h2f  = (_Float16*)alloc((size_t)N_NODES * 32 * 2);
  float*    neigh = (float*)alloc((size_t)N_NODES * 64 * 4);
  float*    csum  = (float*)alloc((size_t)N_NODES * 4);
  float*    vsum  = (float*)alloc((size_t)N_VMS * 4);

  auto packGrid = [](int total) { return dim3((total + 255) / 256); };
  // (src, dst, KA_pad, KA_real, KB_real, Ktot, OUTp, OUTr)
  pack_w<<<packGrid(64 * 192), 256, 0, stream>>>(Wm1, wm1t, 128, 128, 64, 192, 64, 50);
  pack_w<<<packGrid(64 * 192), 256, 0, stream>>>(Wa1, wa1t, 128, 128, 50, 192, 64, 50);
  pack_w<<<packGrid(32 * 128), 256, 0, stream>>>(Wm2, wm2t, 64, 50, 64, 128, 32, 25);
  pack_w<<<packGrid(32 * 96),  256, 0, stream>>>(Wa2, wa2t, 64, 50, 25, 96, 32, 25);
  pack_w<<<packGrid(64 * 96),  256, 0, stream>>>(Wm3, wm3t, 32, 25, 64, 96, 64, 64);
  pack_w<<<packGrid(64 * 96),  256, 0, stream>>>(Wa3, wa3t, 32, 25, 64, 96, 64, 64);
  pack_w<<<packGrid(128 * 64), 256, 0, stream>>>(W1, w1t, 64, 64, 0, 64, 128, 128);
  pack_w<<<packGrid(64 * 128), 256, 0, stream>>>(W2, w2t, 128, 128, 0, 128, 64, 64);

  cvt_f32_f16<<<packGrid(N_NODES * 128), 256, 0, stream>>>(comp, h0f, N_NODES * 128);

  // VM tower (independent of GNN)
  mlp_k<<<dim3((625 + 3) / 4), 128, 0, stream>>>(vm, w1t, b1, w2t, b2, vsum, 625);

  const int edgeTiles = N_EDGES / 16;           // 100000
  const int edgeBlocks = (edgeTiles + 7) / 8;   // 12500
  const int nodeTiles = N_NODES / 16;           // 3125
  const int nodeBlocks = (nodeTiles + 7) / 8;   // 391

  // ---- layer 1: 128 -> 50 (pad 64)
  hipMemsetAsync(neigh, 0, (size_t)N_NODES * 64 * 4, stream);
  edge_msg<128, 64, 50><<<edgeBlocks, 256, 0, stream>>>(
      h0f, ef, src, dst, wm1t, bm1, neigh, edgeTiles);
  apply_k<128, 64, 64, 50, true, false><<<nodeBlocks, 256, 0, stream>>>(
      h0f, neigh, wa1t, ba1, h1f, nullptr, nodeTiles);

  // ---- layer 2: 50(pad64) -> 25 (pad 32)
  hipMemsetAsync(neigh, 0, (size_t)N_NODES * 32 * 4, stream);
  edge_msg<64, 32, 25><<<edgeBlocks, 256, 0, stream>>>(
      h1f, ef, src, dst, wm2t, bm2, neigh, edgeTiles);
  apply_k<64, 32, 32, 25, true, false><<<nodeBlocks, 256, 0, stream>>>(
      h1f, neigh, wa2t, ba2, h2f, nullptr, nodeTiles);

  // ---- layer 3: 25(pad32) -> 64, produce per-node row sums only
  hipMemsetAsync(neigh, 0, (size_t)N_NODES * 64 * 4, stream);
  edge_msg<32, 64, 64><<<edgeBlocks, 256, 0, stream>>>(
      h2f, ef, src, dst, wm3t, bm3, neigh, edgeTiles);
  apply_k<32, 64, 64, 64, false, true><<<nodeBlocks, 256, 0, stream>>>(
      h2f, neigh, wa3t, ba3, nullptr, csum, nodeTiles);

  // ---- scoring
  score_k<<<packGrid(out_size), 256, 0, stream>>>(
      csum, vsum, ecomp, evm, (float*)d_out, out_size);
}
